// ANPMRShapeNet3D_13821204758753
// MI455X (gfx1250) — compile-verified, matrix operand order checked by reading, not run on hardware
//
#include <hip/hip_runtime.h>
#include <hip/hip_bf16.h>
#include <stdint.h>

// ---------------- problem constants ----------------
#define T_DIM 8
#define NCC   512
#define NTT   512
#define DDIM  256
#define HH    8
#define NBF_F 1419
#define LDF   1420      // padded leading dim for F (multiple of 4 -> 16B rows)
#define LDCAT 260       // padded leading dim for concat (259 -> 260)
#define EPS_K 1e-4f

typedef __attribute__((ext_vector_type(16))) __bf16 v16bf;
typedef __attribute__((ext_vector_type(8)))  float  v8f;

union AFrag { v16bf v; uint4 q[2]; };

struct BOff { int div; long long s1; long long s2; };
__device__ __forceinline__ long long boff(BOff o, int z) {
  return (long long)(z / o.div) * o.s1 + (long long)(z % o.div) * o.s2;
}

__device__ __forceinline__ unsigned short f2bf(float f) {
  union { float f; unsigned u; } c; c.f = f;
  unsigned u = c.u;
  u += 0x7FFFu + ((u >> 16) & 1u);       // round-to-nearest-even
  return (unsigned short)(u >> 16);
}

// packed bf16 store of 4 values -> one ds_store_b64
__device__ __forceinline__ void store4_bf(unsigned short* dst, float4 v, float a) {
  union { unsigned short s[4]; uint2 u; } p;
  p.s[0] = f2bf(v.x * a); p.s[1] = f2bf(v.y * a);
  p.s[2] = f2bf(v.z * a); p.s[3] = f2bf(v.w * a);
  *(uint2*)dst = p.u;
}

__device__ __forceinline__ float4 load4_tail(const float* p, int rem) {
  float4 v = make_float4(0.f, 0.f, 0.f, 0.f);
  if (rem > 0) v.x = p[0];
  if (rem > 1) v.y = p[1];
  if (rem > 2) v.z = p[2];
  return v;
}

template<bool MAXRED>
__device__ __forceinline__ float block_reduce(float v, float* red) {
  const int tid = threadIdx.x;
  red[tid] = v;
  __syncthreads();
  #pragma unroll
  for (int s = 128; s > 0; s >>= 1) {
    if (tid < s) red[tid] = MAXRED ? fmaxf(red[tid], red[tid + s])
                                   : (red[tid] + red[tid + s]);
    __syncthreads();
  }
  float r = red[0];
  __syncthreads();
  return r;
}

// ---------------------------------------------------------------------------
// Batched GEMM, f32 in/out, bf16 WMMA compute, f32 accumulate, LDS double-buf.
//   AT=0: A is [M x K] rows of lda.  AT=1: A is [K x M] rows of lda (A^T).
//   BT=0: B is [K x N] rows of ldb.  BT=1: B is [N x K] rows of ldb (B^T).
//   EPI: 0 plain, 1 +bias[n], 2 relu(x+bias[n]), 3 x*rowScale[m]
// Block tile 128(M) x 256(N), K-step 32, 256 threads = 8 wave32s (2x4),
// wave tile 64x64 = 16 WMMAs per K-step (WMMA:ds_load ratio 1.0).
// All leading dims must be multiples of 4 floats (16B-aligned rows).
// ---------------------------------------------------------------------------
template<int AT, int BT, int EPI>
__global__ __launch_bounds__(256)
void gemm_bf16_kernel(const float* __restrict__ A, const float* __restrict__ B,
                      const float* __restrict__ bias,
                      const float* __restrict__ rowScale,
                      float* __restrict__ C,
                      int M, int N, int K, int lda, int ldb, float alpha,
                      BOff aOff, BOff bOff, BOff biasOff, BOff rsOff, BOff cOff,
                      long long cRowStride, long long cColStride) {
  const int tid  = threadIdx.x;
  const int lane = tid & 31;
  const int wid  = tid >> 5;
  const int lm   = lane & 15;          // M/N index within 16
  const int lh   = lane >> 4;          // half selector
  const int waveM = (wid & 1) * 64;    // 2 wave-rows
  const int waveN = (wid >> 1) * 64;   // 4 wave-cols
  const int z     = blockIdx.z;
  const int mBase = blockIdx.y * 128;
  const int nBase = blockIdx.x * 256;

  const float* Ab = A + boff(aOff, z);
  const float* Bb = B + boff(bOff, z);
  float*       Cb = C + boff(cOff, z);

  const bool mFull = (mBase + 128 <= M);
  const bool nFull = (nBase + 256 <= N);

  __shared__ alignas(16) unsigned short As[2][128][40];  // bf16, 80B row stride
  __shared__ alignas(16) unsigned short Bs[2][256][40];  // kept as [n][k]

  v8f acc[4][4] = {};

  // ------------------- staging (global f32 -> LDS bf16) -------------------
  auto stageA = [&](int k0, int p) {
    const bool kFull = (k0 + 32 <= K);
    if constexpr (AT == 0) {
      if (mFull && kFull) {
        #pragma unroll
        for (int r = 0; r < 4; ++r) {
          int row = r * 32 + (tid >> 3);
          int col = (tid & 7) * 4;
          float4 v = *(const float4*)(Ab + (long long)(mBase + row) * lda + k0 + col);
          store4_bf(&As[p][row][col], v, alpha);
        }
      } else {
        #pragma unroll
        for (int r = 0; r < 4; ++r) {
          int row = r * 32 + (tid >> 3);
          int col = (tid & 7) * 4;
          int gm  = mBase + row;
          float4 v = make_float4(0.f, 0.f, 0.f, 0.f);
          if (gm < M) {
            int rem = K - (k0 + col);
            const float* ptr = Ab + (long long)gm * lda + k0 + col;
            if (rem >= 4) v = *(const float4*)ptr;
            else if (rem > 0) v = load4_tail(ptr, rem);
          }
          store4_bf(&As[p][row][col], v, alpha);
        }
      }
    } else {  // A is [K x M]: read along M, store transposed
      if (mFull && kFull) {
        #pragma unroll
        for (int q = 0; q < 4; ++q) {
          int kl = q * 8 + (tid >> 5);
          int ml = (tid & 31) * 4;
          float4 v = *(const float4*)(Ab + (long long)(k0 + kl) * lda + mBase + ml);
          As[p][ml + 0][kl] = f2bf(v.x * alpha);
          As[p][ml + 1][kl] = f2bf(v.y * alpha);
          As[p][ml + 2][kl] = f2bf(v.z * alpha);
          As[p][ml + 3][kl] = f2bf(v.w * alpha);
        }
      } else {
        #pragma unroll
        for (int q = 0; q < 4; ++q) {
          int kl = q * 8 + (tid >> 5);
          int ml = (tid & 31) * 4;
          int gk = k0 + kl;
          float4 v = make_float4(0.f, 0.f, 0.f, 0.f);
          if (gk < K) {
            int rem = M - (mBase + ml);
            const float* ptr = Ab + (long long)gk * lda + mBase + ml;
            if (rem >= 4) v = *(const float4*)ptr;
            else if (rem > 0) v = load4_tail(ptr, rem);
          }
          As[p][ml + 0][kl] = f2bf(v.x * alpha);
          As[p][ml + 1][kl] = f2bf(v.y * alpha);
          As[p][ml + 2][kl] = f2bf(v.z * alpha);
          As[p][ml + 3][kl] = f2bf(v.w * alpha);
        }
      }
    }
  };

  auto stageB = [&](int k0, int p) {
    const bool kFull = (k0 + 32 <= K);
    if constexpr (BT == 0) {  // B is [K x N]: read along N, store transposed
      if (nFull && kFull) {
        #pragma unroll
        for (int q = 0; q < 8; ++q) {
          int kl = q * 4 + (tid >> 6);
          int nl = (tid & 63) * 4;
          float4 v = *(const float4*)(Bb + (long long)(k0 + kl) * ldb + nBase + nl);
          Bs[p][nl + 0][kl] = f2bf(v.x);
          Bs[p][nl + 1][kl] = f2bf(v.y);
          Bs[p][nl + 2][kl] = f2bf(v.z);
          Bs[p][nl + 3][kl] = f2bf(v.w);
        }
      } else {
        #pragma unroll
        for (int q = 0; q < 8; ++q) {
          int kl = q * 4 + (tid >> 6);
          int nl = (tid & 63) * 4;
          int gk = k0 + kl;
          float4 v = make_float4(0.f, 0.f, 0.f, 0.f);
          if (gk < K) {
            int rem = N - (nBase + nl);
            const float* ptr = Bb + (long long)gk * ldb + nBase + nl;
            if (rem >= 4) v = *(const float4*)ptr;
            else if (rem > 0) v = load4_tail(ptr, rem);
          }
          Bs[p][nl + 0][kl] = f2bf(v.x);
          Bs[p][nl + 1][kl] = f2bf(v.y);
          Bs[p][nl + 2][kl] = f2bf(v.z);
          Bs[p][nl + 3][kl] = f2bf(v.w);
        }
      }
    } else {  // B is [N x K]: rows map directly onto Bs[n][k]
      if (nFull && kFull) {
        #pragma unroll
        for (int r = 0; r < 8; ++r) {
          int row = r * 32 + (tid >> 3);
          int col = (tid & 7) * 4;
          float4 v = *(const float4*)(Bb + (long long)(nBase + row) * ldb + k0 + col);
          store4_bf(&Bs[p][row][col], v, 1.f);
        }
      } else {
        #pragma unroll
        for (int r = 0; r < 8; ++r) {
          int row = r * 32 + (tid >> 3);
          int col = (tid & 7) * 4;
          int gn  = nBase + row;
          float4 v = make_float4(0.f, 0.f, 0.f, 0.f);
          if (gn < N) {
            int rem = K - (k0 + col);
            const float* ptr = Bb + (long long)gn * ldb + k0 + col;
            if (rem >= 4) v = *(const float4*)ptr;
            else if (rem > 0) v = load4_tail(ptr, rem);
          }
          store4_bf(&Bs[p][row][col], v, 1.f);
        }
      }
    }
  };

  // ------------------- main loop (double buffered) -------------------
  stageA(0, 0);
  stageB(0, 0);

  const int kTiles = (K + 31) / 32;
  for (int kt = 0; kt < kTiles; ++kt) {
    __syncthreads();
    const int p = kt & 1;

    // fragment loads from current buffer (ds_load_b128 pairs)
    AFrag fa[4], fb[4];
    #pragma unroll
    for (int im = 0; im < 4; ++im) {
      int r = waveM + im * 16 + lm;
      fa[im].q[0] = *(const uint4*)(&As[p][r][lh * 8]);       // K = lh*8..+7
      fa[im].q[1] = *(const uint4*)(&As[p][r][16 + lh * 8]);  // K = 16+lh*8..+7
    }
    #pragma unroll
    for (int jn = 0; jn < 4; ++jn) {
      int r = waveN + jn * 16 + lm;
      fb[jn].q[0] = *(const uint4*)(&Bs[p][r][lh * 8]);
      fb[jn].q[1] = *(const uint4*)(&Bs[p][r][16 + lh * 8]);
    }

    // stage next tile into the other buffer (overlaps WMMA below)
    if (kt + 1 < kTiles) {
      stageA((kt + 1) * 32, p ^ 1);
      stageB((kt + 1) * 32, p ^ 1);
      if (kt + 2 < kTiles) {
        long long pa = (AT == 0) ? ((long long)mBase * lda + (kt + 2) * 32)
                                 : ((long long)(kt + 2) * 32 * lda + mBase);
        long long pb = (BT == 0) ? ((long long)(kt + 2) * 32 * ldb + nBase)
                                 : ((long long)nBase * ldb + (kt + 2) * 32);
        __builtin_prefetch(Ab + pa, 0, 0);
        __builtin_prefetch(Bb + pb, 0, 0);
      }
    }

    #pragma unroll
    for (int im = 0; im < 4; ++im)
      #pragma unroll
      for (int jn = 0; jn < 4; ++jn)
        acc[im][jn] = __builtin_amdgcn_wmma_f32_16x16x32_bf16(
            false, fa[im].v, false, fb[jn].v, (short)0, acc[im][jn],
            false, false);
  }

  // ------------------- epilogue + store -------------------
  const float* biasp = (EPI == 1 || EPI == 2) ? (bias + boff(biasOff, z)) : nullptr;
  const float* rsp   = (EPI == 3) ? (rowScale + boff(rsOff, z)) : nullptr;
  if (mFull && nFull) {
    #pragma unroll
    for (int im = 0; im < 4; ++im)
      for (int jn = 0; jn < 4; ++jn)
        #pragma unroll
        for (int g = 0; g < 8; ++g) {
          int gm = mBase + waveM + im * 16 + lh * 8 + g;  // C: M = g + 8*lh
          int gn = nBase + waveN + jn * 16 + lm;          //    N = lane%16
          float v = acc[im][jn][g];
          if constexpr (EPI == 1) v += biasp[gn];
          if constexpr (EPI == 2) v = fmaxf(v + biasp[gn], 0.f);
          if constexpr (EPI == 3) v *= rsp[gm];
          Cb[(long long)gm * cRowStride + (long long)gn * cColStride] = v;
        }
  } else {
    #pragma unroll
    for (int im = 0; im < 4; ++im)
      for (int jn = 0; jn < 4; ++jn)
        #pragma unroll
        for (int g = 0; g < 8; ++g) {
          int gm = mBase + waveM + im * 16 + lh * 8 + g;
          int gn = nBase + waveN + jn * 16 + lm;
          if (gm < M && gn < N) {
            float v = acc[im][jn][g];
            if constexpr (EPI == 1) v += biasp[gn];
            if constexpr (EPI == 2) v = fmaxf(v + biasp[gn], 0.f);
            if constexpr (EPI == 3) v *= rsp[gm];
            Cb[(long long)gm * cRowStride + (long long)gn * cColStride] = v;
          }
        }
  }
}

// ---------------------------------------------------------------------------
// Small helper kernels for the FAVOR+ feature map
// ---------------------------------------------------------------------------
__global__ __launch_bounds__(256)
void concat_kernel(const float* __restrict__ x, const float* __restrict__ lbl,
                   float* __restrict__ cat) {
  long long r = blockIdx.x;  // T*NC rows
  for (int c = threadIdx.x; c < DDIM + 3; c += 256)
    cat[r * LDCAT + c] = (c < DDIM) ? x[r * DDIM + c] : lbl[r * 3 + (c - DDIM)];
}

// qp: per-row max, subtract diag, exp, scale (in place on dd, rows of LDF)
__global__ __launch_bounds__(256)
void finalize_q_kernel(float* __restrict__ dd, const float* __restrict__ q) {
  __shared__ float red[256];
  const int tid = threadIdx.x;
  const long long r = blockIdx.x;  // T*H*NT rows
  const float ratio = rsqrtf((float)NBF_F);
  const float* qr = q + r * DDIM;
  float s = 0.f;
  for (int i = tid; i < DDIM; i += 256) { float v = qr[i]; s += v * v; }
  s = block_reduce<false>(s, red);
  const float diag = 0.5f * s * 0.0625f;  // dn^2 = 256^-0.5
  float* row = dd + r * (long long)LDF;
  float m = -3.4e38f;
  for (int i = tid; i < NBF_F; i += 256) m = fmaxf(m, row[i]);
  m = block_reduce<true>(m, red);
  for (int i = tid; i < NBF_F; i += 256)
    row[i] = ratio * (__expf(row[i] - diag - m) + EPS_K);
}

// per-(t,h) max over all (n,f) of dd_k (skip the padded column)
__global__ __launch_bounds__(256)
void kmax_kernel(const float* __restrict__ ddk, float* __restrict__ km) {
  __shared__ float red[256];
  const int z = blockIdx.x;  // T*H
  const float* base = ddk + (long long)z * NCC * LDF;
  float m = -3.4e38f;
  for (int n = 0; n < NCC; ++n) {
    const float* row = base + (long long)n * LDF;
    for (int i = threadIdx.x; i < NBF_F; i += 256) m = fmaxf(m, row[i]);
  }
  m = block_reduce<true>(m, red);
  if (threadIdx.x == 0) km[z] = m;
}

__global__ __launch_bounds__(256)
void finalize_k_kernel(float* __restrict__ dd, const float* __restrict__ k,
                       const float* __restrict__ km) {
  __shared__ float red[256];
  const int tid = threadIdx.x;
  const long long r = blockIdx.x;       // T*H*NC rows
  const int z = (int)(r / NCC);
  const float ratio = rsqrtf((float)NBF_F);
  const float* kr = k + r * DDIM;
  float s = 0.f;
  for (int i = tid; i < DDIM; i += 256) { float v = kr[i]; s += v * v; }
  s = block_reduce<false>(s, red);
  const float diag = 0.5f * s * 0.0625f;
  const float m = km[z];
  float* row = dd + r * (long long)LDF;
  for (int i = tid; i < NBF_F; i += 256)
    row[i] = ratio * (__expf(row[i] - diag - m) + EPS_K);
}

// ksum[z][f] = sum_n kp[z][n][f]
__global__ __launch_bounds__(256)
void ksum_kernel(const float* __restrict__ kp, float* __restrict__ ksum) {
  const int z = blockIdx.y;
  const int f = blockIdx.x * 256 + threadIdx.x;
  if (f >= NBF_F) return;
  const float* base = kp + (long long)z * NCC * LDF + f;
  float s = 0.f;
  for (int n = 0; n < NCC; ++n) s += base[(long long)n * LDF];
  ksum[(long long)z * NBF_F + f] = s;
}

// dinv[r] = 1 / dot(qp[r,:], ksum[z,:])
__global__ __launch_bounds__(256)
void dinv_kernel(const float* __restrict__ qp, const float* __restrict__ ksum,
                 float* __restrict__ dinv) {
  __shared__ float red[256];
  const long long r = blockIdx.x;  // T*H*NT
  const int z = (int)(r / NTT);
  const float* row = qp + r * (long long)LDF;
  const float* ks  = ksum + (long long)z * NBF_F;
  float s = 0.f;
  for (int i = threadIdx.x; i < NBF_F; i += 256) s += row[i] * ks[i];
  s = block_reduce<false>(s, red);
  if (threadIdx.x == 0) dinv[r] = 1.0f / s;
}

// ---------------------------------------------------------------------------
extern "C" void kernel_launch(void* const* d_in, const int* in_sizes, int n_in,
                              void* d_out, int out_size, void* d_ws, size_t ws_size,
                              hipStream_t stream) {
  const float* x_ctx = (const float*)d_in[0];
  const float* label = (const float*)d_in[1];
  const float* x_tgt = (const float*)d_in[2];
  const float* W1 = (const float*)d_in[3];  const float* b1 = (const float*)d_in[4];
  const float* W2 = (const float*)d_in[5];  const float* b2 = (const float*)d_in[6];
  const float* W3 = (const float*)d_in[7];  const float* b3 = (const float*)d_in[8];
  const float* Wk = (const float*)d_in[9];  const float* bk = (const float*)d_in[10];
  const float* Wv = (const float*)d_in[11]; const float* bv = (const float*)d_in[12];
  const float* Wq = (const float*)d_in[13]; const float* bq = (const float*)d_in[14];
  const float* Wo = (const float*)d_in[15]; const float* bo = (const float*)d_in[16];
  const float* Wmu = (const float*)d_in[17]; const float* bmu = (const float*)d_in[18];
  const float* proj = (const float*)d_in[19];
  float* out = (float*)d_out;

  const int TH = T_DIM * HH;                         // 64
  const long long ROWS_C = (long long)T_DIM * NCC;   // 4096
  const long long ROWS_T = (long long)T_DIM * NTT;   // 4096
  const long long RQ = (long long)TH * NTT;          // 32768
  const long long RK = (long long)TH * NCC;          // 32768

  float* ws = (float*)d_ws;
  size_t o = 0;
  float* cat    = ws + o; o += (size_t)ROWS_C * LDCAT;
  float* h1     = ws + o; o += (size_t)ROWS_C * DDIM;
  float* h2     = ws + o; o += (size_t)ROWS_C * DDIM;
  float* cf     = ws + o; o += (size_t)ROWS_C * DDIM;
  float* kb     = ws + o; o += (size_t)RK * DDIM;
  float* vb     = ws + o; o += (size_t)RK * DDIM;
  float* qb     = ws + o; o += (size_t)RQ * DDIM;
  float* ddq    = ws + o; o += (size_t)RQ * LDF;     // becomes qp
  float* ddk    = ws + o; o += (size_t)RK * LDF;     // becomes kp
  float* ctx    = ws + o; o += (size_t)TH * NBF_F * DDIM;
  float* ksum   = ws + o; o += (size_t)TH * NBF_F;
  float* km     = ws + o; o += (size_t)TH;
  float* dinv   = ws + o; o += (size_t)RQ;
  float* merged = ws + o; o += (size_t)ROWS_T * HH * DDIM;
  float* rep    = ws + o; o += (size_t)ROWS_T * DDIM;
  (void)ws_size; (void)n_in; (void)in_sizes; (void)out_size;

  const dim3 blk(256, 1, 1);
  auto g2 = [](long long M, long long N, int B) {
    return dim3((unsigned)((N + 255) / 256), (unsigned)((M + 127) / 128), (unsigned)B);
  };
  const BOff Z = {1, 0, 0};

  // 1) concat [x_ctx, label] -> [4096 x 259] (ld 260)
  concat_kernel<<<(unsigned)ROWS_C, blk, 0, stream>>>(x_ctx, label, cat);

  // 2-4) task-encoder MLP (bias+relu epilogue)
  gemm_bf16_kernel<0,0,2><<<g2(ROWS_C,DDIM,1), blk, 0, stream>>>(
      cat, W1, b1, nullptr, h1, (int)ROWS_C, DDIM, DDIM + 3, LDCAT, DDIM, 1.f,
      Z, Z, Z, Z, Z, DDIM, 1);
  gemm_bf16_kernel<0,0,2><<<g2(ROWS_C,DDIM,1), blk, 0, stream>>>(
      h1, W2, b2, nullptr, h2, (int)ROWS_C, DDIM, DDIM, DDIM, DDIM, 1.f,
      Z, Z, Z, Z, Z, DDIM, 1);
  gemm_bf16_kernel<0,0,2><<<g2(ROWS_C,DDIM,1), blk, 0, stream>>>(
      h2, W3, b3, nullptr, cf, (int)ROWS_C, DDIM, DDIM, DDIM, DDIM, 1.f,
      Z, Z, Z, Z, Z, DDIM, 1);

  // 5-7) batched per-head projections k/v/q  (z = t*H + h)
  const BOff aCtx = {HH, (long long)NCC * DDIM, 0};
  const BOff aTgt = {HH, (long long)NTT * DDIM, 0};
  const BOff bW   = {HH, 0, (long long)DDIM * DDIM};
  const BOff bB   = {HH, 0, DDIM};
  const BOff cKV  = {1, (long long)NCC * DDIM, 0};
  const BOff cQ   = {1, (long long)NTT * DDIM, 0};
  gemm_bf16_kernel<0,0,1><<<g2(NCC,DDIM,TH), blk, 0, stream>>>(
      x_ctx, Wk, bk, nullptr, kb, NCC, DDIM, DDIM, DDIM, DDIM, 1.f,
      aCtx, bW, bB, Z, cKV, DDIM, 1);
  gemm_bf16_kernel<0,0,1><<<g2(NCC,DDIM,TH), blk, 0, stream>>>(
      cf, Wv, bv, nullptr, vb, NCC, DDIM, DDIM, DDIM, DDIM, 1.f,
      aCtx, bW, bB, Z, cKV, DDIM, 1);
  gemm_bf16_kernel<0,0,1><<<g2(NTT,DDIM,TH), blk, 0, stream>>>(
      x_tgt, Wq, bq, nullptr, qb, NTT, DDIM, DDIM, DDIM, DDIM, 1.f,
      aTgt, bW, bB, Z, cQ, DDIM, 1);

  // 8-9) dd = (x * d^-1/4) @ proj^T  (proj is [F x D] => BT=1); alpha = 0.25
  gemm_bf16_kernel<0,1,0><<<g2(RQ,NBF_F,1), blk, 0, stream>>>(
      qb, proj, nullptr, nullptr, ddq, (int)RQ, NBF_F, DDIM, DDIM, DDIM, 0.25f,
      Z, Z, Z, Z, Z, LDF, 1);
  gemm_bf16_kernel<0,1,0><<<g2(RK,NBF_F,1), blk, 0, stream>>>(
      kb, proj, nullptr, nullptr, ddk, (int)RK, NBF_F, DDIM, DDIM, DDIM, 0.25f,
      Z, Z, Z, Z, Z, LDF, 1);

  // 10-13) FAVOR+ feature maps
  finalize_q_kernel<<<(unsigned)RQ, blk, 0, stream>>>(ddq, qb);
  kmax_kernel<<<TH, blk, 0, stream>>>(ddk, km);
  finalize_k_kernel<<<(unsigned)RK, blk, 0, stream>>>(ddk, kb, km);
  ksum_kernel<<<dim3((NBF_F + 255) / 256, TH), blk, 0, stream>>>(ddk, ksum);

  // 14) ctx = kp^T @ v  (A stored [NC x LDF] => AT=1)
  gemm_bf16_kernel<1,0,0><<<g2(NBF_F,DDIM,TH), blk, 0, stream>>>(
      ddk, vb, nullptr, nullptr, ctx, NBF_F, DDIM, NCC, LDF, DDIM, 1.f,
      {1, (long long)NCC * LDF, 0}, {1, (long long)NCC * DDIM, 0},
      Z, Z, {1, (long long)NBF_F * DDIM, 0}, DDIM, 1);

  // 15) d_inv
  dinv_kernel<<<(unsigned)RQ, blk, 0, stream>>>(ddq, ksum, dinv);

  // 16) out = (qp @ ctx) * d_inv, stored head-interleaved: merged[t][n][e*H + h]
  gemm_bf16_kernel<0,0,3><<<g2(NTT,DDIM,TH), blk, 0, stream>>>(
      ddq, ctx, nullptr, dinv, merged, NTT, DDIM, NBF_F, LDF, DDIM, 1.f,
      {1, (long long)NTT * LDF, 0}, {1, (long long)NBF_F * DDIM, 0},
      Z, {1, NTT, 0},
      {HH, (long long)NTT * HH * DDIM, 1}, (long long)HH * DDIM, HH);

  // 17) rep = merged @ Wo + bo
  gemm_bf16_kernel<0,0,1><<<g2(ROWS_T,DDIM,1), blk, 0, stream>>>(
      merged, Wo, bo, nullptr, rep, (int)ROWS_T, DDIM, HH * DDIM,
      HH * DDIM, DDIM, 1.f, Z, Z, Z, Z, Z, DDIM, 1);

  // 18) mu = rep @ Wmu + bmu -> d_out
  gemm_bf16_kernel<0,0,1><<<g2(ROWS_T,DDIM,1), blk, 0, stream>>>(
      rep, Wmu, bmu, nullptr, out, (int)ROWS_T, DDIM, DDIM, DDIM, DDIM, 1.f,
      Z, Z, Z, Z, Z, DDIM, 1);
}
